// MultiFea_Block_70970039599533
// MI455X (gfx1250) — 8-bit matrix operand order repaired and checked by reading, compiled, not run on hardware
//
#include <hip/hip_runtime.h>

typedef __attribute__((ext_vector_type(8))) int v8i;

#define M_TOTAL 50176
#define KCH 36
#define BM 64

static __device__ __forceinline__ signed char sgn8(float v) {
    return (signed char)((v > 0.f) ? 1 : ((v < 0.f) ? -1 : 0));
}

__global__ __launch_bounds__(256) void k_init_stats(int* s1, unsigned long long* q1,
                                                    int* s2, unsigned long long* q2) {
    const int t = threadIdx.x;
    s1[t] = 0; q1[t] = 0ull; s2[t] = 0; q2[t] = 0ull;
}

__global__ __launch_bounds__(256) void k_actprep(const float* __restrict__ x,
                                                 signed char* __restrict__ xb) {
    const int id = blockIdx.x * 256 + threadIdx.x;
    const int hw = id % 784;
    const int rest = id / 784;
    const int c = rest & 255;
    const int n = rest >> 8;
    xb[((size_t)(n * 784 + hw)) * 256 + c] = sgn8(x[id]);
}

__global__ __launch_bounds__(256) void k_wprep(const float* __restrict__ w,
                                               signed char* __restrict__ wf) {
    const int id = blockIdx.x * 256 + threadIdx.x;
    const int i    = id & 31;
    const int lane = (id >> 5) & 31;
    const int ns   = (id >> 10) & 15;
    const int kc   = id >> 14;
    const int klocal = (i & 7) + ((i >> 3) << 4) + ((lane >> 4) << 3);
    const int k = kc * 64 + klocal;
    const int tap = k >> 8;
    const int c = k & 255;
    const int cout = ns * 16 + (lane & 15);
    wf[id] = sgn8(w[((size_t)cout * 256 + c) * 9 + tap]);
}

__global__ __launch_bounds__(256) void k_bconv(
    const signed char* __restrict__ A,
    const signed char* __restrict__ BF,
    short* __restrict__ Y,
    int* __restrict__ gSum, unsigned long long* __restrict__ gSq)
{
    __shared__ __align__(32) signed char lds_a[2][BM * 64];
    __shared__ __align__(32) signed char lds_b[2][16384];
    __shared__ int sSum[256];
    __shared__ int sSq[256];

    const int t = threadIdx.x;
    const int wid = t >> 5;
    const int lane = t & 31;
    const int mBase = blockIdx.x * BM;

    sSum[t] = 0; sSq[t] = 0;

    v8i acc[8];
#pragma unroll
    for (int j = 0; j < 8; ++j)
#pragma unroll
        for (int e = 0; e < 8; ++e) acc[j][e] = 0;

    const int r = t >> 2;
    const int q = t & 3;
    const int m = mBase + r;
    const int nImg = m / 784;
    const int hw = m - nImg * 784;
    const int h = hw / 28;
    const int w = hw - h * 28;

    const int msub = wid >> 1;
    const int nhalf = wid & 1;
    const int lh = lane >> 4;
    const int ml = lane & 15;

    auto loadA = [&](int kc, unsigned long long& L0, unsigned long long& L1) {
        const int tap = kc >> 2;
        const int ky = tap / 3;
        const int kx = tap - ky * 3;
        const int hs = h + ky - 1;
        const int ws2 = w + kx - 1;
        L0 = 0ull; L1 = 0ull;
        if (hs >= 0 && hs < 28 && ws2 >= 0 && ws2 < 28) {
            const size_t ms = (size_t)(nImg * 784 + hs * 28 + ws2);
            const int c0 = ((kc & 3) << 6) + (q << 4);
            const unsigned long long* p =
                (const unsigned long long*)(A + ms * 256 + c0);
            L0 = p[0]; L1 = p[1];
        }
    };
    auto issueB = [&](int kc, int buf) {
        const signed char* src = BF + (size_t)kc * 16384 + t * 16;
        signed char* dst = &lds_b[buf][t * 16];
#pragma unroll
        for (int p = 0; p < 4; ++p) {
            unsigned int loff = (unsigned int)(size_t)(dst + p * 4096);
            unsigned long long g = (unsigned long long)(size_t)(src + p * 4096);
            asm volatile("global_load_async_to_lds_b128 %0, %1, off"
                         :: "v"(loff), "v"(g) : "memory");
        }
    };

    unsigned long long a0, a1;
    loadA(0, a0, a1);
    issueB(0, 0);

    for (int kc = 0; kc < KCH; ++kc) {
        const int cur = kc & 1;
        *(unsigned long long*)(&lds_a[cur][r * 64 + (q << 3)]) = a0;
        *(unsigned long long*)(&lds_a[cur][r * 64 + 32 + (q << 3)]) = a1;
        if (kc + 1 < KCH) {
            loadA(kc + 1, a0, a1);
            issueB(kc + 1, cur ^ 1);
            asm volatile("s_wait_asynccnt 0x4" ::: "memory");
        } else {
            asm volatile("s_wait_asynccnt 0x0" ::: "memory");
        }
        __syncthreads();
        const v8i af = *(const v8i*)(&lds_a[cur][(msub * 16 + ml) * 64 + lh * 32]);
        v8i bf[8];
#pragma unroll
        for (int j = 0; j < 8; ++j)
            bf[j] = *(const v8i*)(&lds_b[cur][((nhalf * 8 + j) * 32 + lane) * 32]);
#pragma unroll
        for (int j = 0; j < 8; ++j)
            acc[j] = __builtin_amdgcn_wmma_i32_16x16x64_iu8(
                true, af, true, bf[j], acc[j], false, false);
        __syncthreads();
    }
#pragma unroll
    for (int j = 0; j < 8; ++j) {
        const int col = nhalf * 128 + j * 16 + ml;
        int s = 0, sq = 0;
#pragma unroll
        for (int rr = 0; rr < 8; ++rr) {
            const int v = acc[j][rr];
            const int row = msub * 16 + rr + 8 * lh;
            Y[(size_t)(mBase + row) * 256 + col] = (short)v;
            s += v; sq += v * v;
        }
        atomicAdd(&sSum[col], s);
        atomicAdd(&sSq[col], sq);
    }
    __syncthreads();
    atomicAdd(&gSum[t], sSum[t]);
    atomicAdd(&gSq[t], (unsigned long long)(unsigned int)sSq[t]);
}

__global__ __launch_bounds__(256) void k_stats(const int* __restrict__ gSum,
                                               const unsigned long long* __restrict__ gSq,
                                               float* __restrict__ mean,
                                               float* __restrict__ rstd) {
    const int c = threadIdx.x;
    const double mu = (double)gSum[c] / (double)M_TOTAL;
    const double ex2 = (double)gSq[c] / (double)M_TOTAL;
    const double var = ex2 - mu * mu;
    mean[c] = (float)mu;
    rstd[c] = rsqrtf((float)var + 1e-5f);
}

__global__ __launch_bounds__(256) void k_ew1(
    const short* __restrict__ Y, const float* __restrict__ x,
    const float* __restrict__ gamma, const float* __restrict__ beta,
    const float* __restrict__ mean, const float* __restrict__ rstd,
    float* __restrict__ out1, signed char* __restrict__ a2)
{
    __shared__ float xt[32][33];
    const int t = threadIdx.x;
    const int tx = t & 31, ty = t >> 5;
    const int hw0 = blockIdx.x * 32, c0 = blockIdx.y * 32, n = blockIdx.z;
#pragma unroll
    for (int p = 0; p < 4; ++p) {
        const int c = c0 + ty + p * 8;
        const int hw = hw0 + tx;
        xt[ty + p * 8][tx] = (hw < 784) ? x[((size_t)n * 256 + c) * 784 + hw] : 0.f;
    }
    __syncthreads();
#pragma unroll
    for (int p = 0; p < 4; ++p) {
        const int hw = hw0 + ty + p * 8;
        const int c = c0 + tx;
        if (hw < 784) {
            const size_t idx = (size_t)(n * 784 + hw) * 256 + c;
            const float yv = (float)Y[idx];
            const float bn = (yv - mean[c]) * rstd[c] * gamma[c] + beta[c];
            float v = bn + xt[tx][ty + p * 8];
            v = fminf(1.f, fmaxf(-1.f, v));
            out1[idx] = v;
            a2[idx] = sgn8(v);
        }
    }
}

__global__ __launch_bounds__(256) void k_final(
    const short* __restrict__ Y, const float* __restrict__ out1,
    const float* __restrict__ gamma, const float* __restrict__ beta,
    const float* __restrict__ mean, const float* __restrict__ rstd,
    float* __restrict__ dout)
{
    __shared__ float tt[32][33];
    const int t = threadIdx.x;
    const int tx = t & 31, ty = t >> 5;
    const int hw0 = blockIdx.x * 32, c0 = blockIdx.y * 32, n = blockIdx.z;
#pragma unroll
    for (int p = 0; p < 4; ++p) {
        const int hw = hw0 + ty + p * 8;
        const int c = c0 + tx;
        float v = 0.f;
        if (hw < 784) {
            const size_t idx = (size_t)(n * 784 + hw) * 256 + c;
            const float yv = (float)Y[idx];
            const float bn = (yv - mean[c]) * rstd[c] * gamma[c] + beta[c];
            v = bn + out1[idx];
            v = fminf(1.f, fmaxf(-1.f, v));
        }
        tt[tx][ty + p * 8] = v;
    }
    __syncthreads();
#pragma unroll
    for (int p = 0; p < 4; ++p) {
        const int c = c0 + ty + p * 8;
        const int hw = hw0 + tx;
        if (hw < 784) dout[((size_t)n * 256 + c) * 784 + hw] = tt[ty + p * 8][tx];
    }
}

extern "C" void kernel_launch(void* const* d_in, const int* in_sizes, int n_in,
                              void* d_out, int out_size, void* d_ws, size_t ws_size,
                              hipStream_t stream) {
    const float* x  = (const float*)d_in[0];
    const float* w1 = (const float*)d_in[1];
    const float* g1 = (const float*)d_in[2];
    const float* b1 = (const float*)d_in[3];
    const float* w2 = (const float*)d_in[4];
    const float* g2 = (const float*)d_in[5];
    const float* b2 = (const float*)d_in[6];
    float* out = (float*)d_out;

    char* ws = (char*)d_ws;
    const size_t OFF_XB  = 0;
    const size_t OFF_A2  = OFF_XB + 12845056;
    const size_t OFF_WF1 = OFF_A2 + 12845056;
    const size_t OFF_WF2 = OFF_WF1 + 589824;
    const size_t OFF_Y   = OFF_WF2 + 589824;
    const size_t OFF_O1  = OFF_Y + 25690112;
    const size_t OFF_ST  = OFF_O1 + 51380224;
    const size_t NEED    = OFF_ST + 10240;
    if (ws_size < NEED) return;

    signed char* XB  = (signed char*)(ws + OFF_XB);
    signed char* A2  = (signed char*)(ws + OFF_A2);
    signed char* WF1 = (signed char*)(ws + OFF_WF1);
    signed char* WF2 = (signed char*)(ws + OFF_WF2);
    short* Yb = (short*)(ws + OFF_Y);
    float* O1 = (float*)(ws + OFF_O1);
    int* gSum1 = (int*)(ws + OFF_ST);
    unsigned long long* gSq1 = (unsigned long long*)(ws + OFF_ST + 1024);
    int* gSum2 = (int*)(ws + OFF_ST + 3072);
    unsigned long long* gSq2 = (unsigned long long*)(ws + OFF_ST + 4096);
    float* mean1 = (float*)(ws + OFF_ST + 6144);
    float* rstd1 = (float*)(ws + OFF_ST + 7168);
    float* mean2 = (float*)(ws + OFF_ST + 8192);
    float* rstd2 = (float*)(ws + OFF_ST + 9216);

    k_init_stats<<<1, 256, 0, stream>>>(gSum1, gSq1, gSum2, gSq2);
    k_actprep<<<50176, 256, 0, stream>>>(x, XB);
    k_wprep<<<2304, 256, 0, stream>>>(w1, WF1);
    k_wprep<<<2304, 256, 0, stream>>>(w2, WF2);

    k_bconv<<<784, 256, 0, stream>>>(XB, WF1, Yb, gSum1, gSq1);
    k_stats<<<1, 256, 0, stream>>>(gSum1, gSq1, mean1, rstd1);
    k_ew1<<<dim3(25, 8, 64), 256, 0, stream>>>(Yb, x, g1, b1, mean1, rstd1, O1, A2);

    k_bconv<<<784, 256, 0, stream>>>(A2, WF2, Yb, gSum2, gSq2);
    k_stats<<<1, 256, 0, stream>>>(gSum2, gSq2, mean2, rstd2);
    k_final<<<dim3(25, 8, 64), 256, 0, stream>>>(Yb, O1, g2, b2, mean2, rstd2, out);

    (void)in_sizes; (void)n_in; (void)out_size;
}
